// MultiHeadedAttentionBlock_69853348102901
// MI455X (gfx1250) — matrix operand order repaired and checked by reading, not compiled, not run on hardware
//
#include <hip/hip_runtime.h>
#include <hip/hip_fp16.h>

// ---------------------------------------------------------------------------
// MI455X (gfx1250, wave32) fused multi-head attention block.
//   kernel 1: kqv = x @ Wkqv + b, fused RoPE(q,k), store f16 [3][B][H][T][dk]
//   kernel 2: flash attention per (b,h), 128-row query tiles, f16 WMMA
//   kernel 3: out = ctx @ Wo + bo, fp32 output
// Matrix math via v_wmma_f32_16x16x32_f16 (f32 accumulate); f16 tile staging
// uses GLOBAL_LOAD_ASYNC_TO_LDS_B128 (ASYNCcnt) when the builtin is available.
// ---------------------------------------------------------------------------

typedef _Float16 half_t;
typedef _Float16 v16h __attribute__((ext_vector_type(16)));
typedef _Float16 v8h  __attribute__((ext_vector_type(8)));
typedef float    v8f  __attribute__((ext_vector_type(8)));

#define B_   2
#define T_   2048
#define D_   2048
#define H_   16
#define DK_  128
#define ND_  6144                         // 3*D
static constexpr size_t SEC_ = (size_t)B_ * H_ * T_ * DK_;   // halves per k/q/v section

// ---- async global->LDS staging (gfx1250 ASYNCcnt path), guarded -----------
#if defined(__gfx1250__) && __has_builtin(__builtin_amdgcn_global_load_async_to_lds_b128)
#define HAVE_ASYNC_LDS 1
// Builtin signature (from hipcc diagnostic): param0 = addrspace(1) int4*,
// param1 = addrspace(3) int4*, then imm offset, imm cpol.
typedef int v4i_gcc __attribute__((vector_size(16)));
typedef __attribute__((address_space(1))) v4i_gcc gas_v4i;
typedef __attribute__((address_space(3))) v4i_gcc las_v4i;
__device__ __forceinline__ void async_copy16B(const half_t* g, half_t* l) {
  __builtin_amdgcn_global_load_async_to_lds_b128((gas_v4i*)g, (las_v4i*)l, 0, 0);
}
__device__ __forceinline__ void async_wait0() {
#if __has_builtin(__builtin_amdgcn_s_wait_asynccnt)
  __builtin_amdgcn_s_wait_asynccnt(0);
#else
  asm volatile("s_wait_asynccnt 0" ::: "memory");
#endif
}
#endif

__device__ __forceinline__ v8f zero8() {
  v8f r;
#pragma unroll
  for (int i = 0; i < 8; ++i) r[i] = 0.0f;
  return r;
}

__device__ __forceinline__ v16h combine16(v8h lo, v8h hi) {
  v16h r;
#pragma unroll
  for (int i = 0; i < 8; ++i) { r[i] = lo[i]; r[i + 8] = hi[i]; }
  return r;
}

__device__ __forceinline__ v8f wmma_f16(v16h a, v16h b, v8f c) {
  // (neg_a, A, neg_b, B, c_mod, C, reuse_a, reuse_b)
  return __builtin_amdgcn_wmma_f32_16x16x32_f16(false, a, false, b, (short)0, c,
                                                false, false);
}

// A-fragment 16x32 f16 (gfx1250 layout): lane L holds row M=L&15;
// halves 0..7 -> K=(L>>4)*8+j, halves 8..15 -> K=16+(L>>4)*8+(j-8).
__device__ __forceinline__ v16h load_afrag(const half_t* base, int ld, int lane) {
  const int row = lane & 15;
  const int kb  = (lane >> 4) * 8;
  const v8h* lo = (const v8h*)(base + row * ld + kb);
  const v8h* hi = (const v8h*)(base + row * ld + 16 + kb);
  return combine16(*lo, *hi);
}

// B-fragment 32x16 f16: lane L holds col N=L&15; halves 0..7 -> K=(L>>4)*8+j, 8..15 -> K=16+(L>>4)*8+(j-8).
// Storage is K-contiguous per column: base[col*ld + k].
__device__ __forceinline__ v16h load_bfrag(const half_t* base, int ld, int lane) {
  const int col = lane & 15;
  const int kb  = (lane >> 4) * 8;
  const v8h* lo = (const v8h*)(base + col * ld + kb);
  const v8h* hi = (const v8h*)(base + col * ld + kb + 16);
  return combine16(*lo, *hi);
}

// ---------------------------------------------------------------------------
// Kernel 1: QKV projection + bias + RoPE, f16 output [sec][B][H][T][dk]
// grid: (ND/128, B*T/128), block: 256 (8 waves); tile 128x128, K-step 32
// ---------------------------------------------------------------------------
__global__ __launch_bounds__(256) void qkv_rope_kernel(
    const float* __restrict__ x, const float* __restrict__ Wkqv,
    const float* __restrict__ bkqv, half_t* __restrict__ qkv) {
  __shared__ half_t As[128 * 32];   // A tile, row-major [m][k]
  __shared__ half_t Bs[128 * 32];   // B tile transposed [n][k]

  const int n0 = blockIdx.x * 128;
  const int m0 = blockIdx.y * 128;
  const int tid  = threadIdx.x;
  const int lane = tid & 31;
  const int w    = tid >> 5;
  const int wm = (w & 3) * 32;      // wave covers 32 rows  (2 m-tiles)
  const int wn = (w >> 2) * 64;     // wave covers 64 cols  (4 n-tiles)

  v8f acc[2][4];
#pragma unroll
  for (int i = 0; i < 2; ++i)
#pragma unroll
    for (int j = 0; j < 4; ++j) acc[i][j] = zero8();

  const int arow = tid >> 1, ac0 = (tid & 1) * 16;   // A stage: 2 thr/row
  const int brow = tid >> 3, bc0 = (tid & 7) * 16;   // B stage: 8 thr/row

  for (int kk = 0; kk < D_; kk += 32) {
    // ---- stage A: x[m0..m0+127][kk..kk+31] -> f16 LDS
    {
      const float4* src = (const float4*)(x + (size_t)(m0 + arow) * D_ + kk + ac0);
      half_t* dst = As + arow * 32 + ac0;
#pragma unroll
      for (int i = 0; i < 4; ++i) {
        float4 v = src[i];
        dst[i * 4 + 0] = (half_t)v.x; dst[i * 4 + 1] = (half_t)v.y;
        dst[i * 4 + 2] = (half_t)v.z; dst[i * 4 + 3] = (half_t)v.w;
      }
    }
    // ---- stage B transposed: Wkqv[kk..+31][n0..+127] -> Bs[n][k]
    {
      const float4* src = (const float4*)(Wkqv + (size_t)(kk + brow) * ND_ + n0 + bc0);
#pragma unroll
      for (int i = 0; i < 4; ++i) {
        float4 v = src[i];
        Bs[(bc0 + i * 4 + 0) * 32 + brow] = (half_t)v.x;
        Bs[(bc0 + i * 4 + 1) * 32 + brow] = (half_t)v.y;
        Bs[(bc0 + i * 4 + 2) * 32 + brow] = (half_t)v.z;
        Bs[(bc0 + i * 4 + 3) * 32 + brow] = (half_t)v.w;
      }
    }
    // prefetch next K-step tiles into cache (global_prefetch_b8)
    if (kk + 32 < D_) {
      __builtin_prefetch(x + (size_t)(m0 + arow) * D_ + kk + 32 + ac0, 0, 1);
      __builtin_prefetch(Wkqv + (size_t)(kk + 32 + brow) * ND_ + n0 + bc0, 0, 1);
    }
    __syncthreads();

    v16h a0 = load_afrag(As + (wm +  0) * 32, 32, lane);
    v16h a1 = load_afrag(As + (wm + 16) * 32, 32, lane);
#pragma unroll
    for (int nt = 0; nt < 4; ++nt) {
      v16h b = load_bfrag(Bs + (wn + nt * 16) * 32, 32, lane);
      acc[0][nt] = wmma_f16(a0, b, acc[0][nt]);
      acc[1][nt] = wmma_f16(a1, b, acc[1][nt]);
    }
    __syncthreads();
  }

  // ---- epilogue: bias + RoPE (RoPE partner lives in lane^1) + f16 store
  const float LN_BASE = 9.210340371976184f;  // ln(10000)
#pragma unroll
  for (int mt = 0; mt < 2; ++mt) {
#pragma unroll
    for (int nt = 0; nt < 4; ++nt) {
      const int n_g  = n0 + wn + nt * 16 + (lane & 15);
      const float bias = bkqv[n_g];
      const int sec  = n_g >> 11;        // 0=k 1=q 2=v (block-uniform)
      const int col  = n_g & (D_ - 1);
      const int h    = col >> 7;
      const int dcol = col & (DK_ - 1);
      const int m_base = m0 + wm + mt * 16 + 8 * (lane >> 4);
#pragma unroll
      for (int r = 0; r < 8; ++r) {
        const int m_g = m_base + r;
        const int t   = m_g & (T_ - 1);
        const int bb  = m_g >> 11;
        float val = acc[mt][nt][r] + bias;
        float partner = __shfl_xor(val, 1, 32);
        float outv = val;
        if (sec < 2) {  // RoPE on k and q
          const float fi   = (float)(dcol >> 1);
          const float invf = __expf(-(2.0f * fi / (float)DK_) * LN_BASE);
          const float ang  = (float)t * invf;
          float s, c;
          __sincosf(ang, &s, &c);
          outv = (dcol & 1) ? (partner * s + val * c) : (val * c - partner * s);
        }
        const size_t idx = (size_t)sec * SEC_ +
            ((((size_t)bb * H_ + h) * T_ + t) * DK_ + dcol);
        qkv[idx] = (half_t)outv;
      }
    }
  }
}

// ---------------------------------------------------------------------------
// Kernel 2: flash attention per (b,h); grid (T/128, B*H), block 256.
// Wave w owns query rows [qt*128 + w*16, +16).  LDS: K tile + V^T tile (64KB);
// K-tile region is reused as per-wave P scratch for the C->A layout shuffle.
// K tile staged with GLOBAL_LOAD_ASYNC_TO_LDS_B128 when available.
// ---------------------------------------------------------------------------
__global__ __launch_bounds__(256) void attn_kernel(
    const half_t* __restrict__ qkv, half_t* __restrict__ ctx) {
  __shared__ half_t kls[128 * 128];   // k[token][d]; later P scratch (8 x 16x128)
  __shared__ half_t vls[128 * 128];   // v^T[d][token]

  const int qt = blockIdx.x;          // query tile
  const int bh = blockIdx.y;
  const int b = bh >> 4, h = bh & 15;
  const int tid = threadIdx.x, lane = tid & 31, w = tid >> 5;

  const size_t head_off = (((size_t)b * H_ + h) * T_) * DK_;
  const half_t* kptr = qkv + head_off;               // section 0 = k
  const half_t* qptr = qkv + SEC_ + head_off;        // section 1 = q
  const half_t* vptr = qkv + 2 * SEC_ + head_off;    // section 2 = v

  // persistent q A-fragments (16 rows x 128 dk = 4 K-chunks)
  v16h aq[4];
#pragma unroll
  for (int c = 0; c < 4; ++c)
    aq[c] = load_afrag(qptr + (size_t)(qt * 128 + w * 16) * DK_ + c * 32, DK_, lane);

  v8f o[8];
#pragma unroll
  for (int i = 0; i < 8; ++i) o[i] = zero8();
  float mrow[8], lrow[8];
#pragma unroll
  for (int r = 0; r < 8; ++r) { mrow[r] = -1e30f; lrow[r] = 0.0f; }

  const int stok = tid >> 1, sc0 = (tid & 1) * 64;   // staging: 2 thr/token

  for (int jt = 0; jt <= qt; ++jt) {
    __syncthreads();  // previous iter's LDS readers are done
    // ---- stage K tile (async DMA to LDS when available) --------------------
    {
      const half_t* ks = kptr + (size_t)(jt * 128 + stok) * DK_ + sc0;
      half_t* kd = kls + stok * 128 + sc0;
#ifdef HAVE_ASYNC_LDS
#pragma unroll
      for (int i = 0; i < 8; ++i) async_copy16B(ks + i * 8, kd + i * 8);
#else
      const v8h* ksv = (const v8h*)ks;
      v8h* kdv = (v8h*)kd;
#pragma unroll
      for (int i = 0; i < 8; ++i) kdv[i] = ksv[i];
#endif
      // ---- stage V tile transposed (needs VALU shuffle, manual path)
      const v8h* vs = (const v8h*)(vptr + (size_t)(jt * 128 + stok) * DK_ + sc0);
#pragma unroll
      for (int i = 0; i < 8; ++i) {
        v8h vv = vs[i];
#pragma unroll
        for (int e = 0; e < 8; ++e) vls[(sc0 + i * 8 + e) * 128 + stok] = vv[e];
      }
    }
#ifdef HAVE_ASYNC_LDS
    async_wait0();     // drain our wave's ASYNCcnt before the barrier
#endif
    __syncthreads();

    // ---- S = q @ k^T : 8 n-tiles x 4 K-chunks
    v8f sacc[8];
#pragma unroll
    for (int nt = 0; nt < 8; ++nt) {
      v8f s = zero8();
#pragma unroll
      for (int c = 0; c < 4; ++c) {
        v16h bk = load_bfrag(kls + (nt * 16) * 128 + c * 32, 128, lane);
        s = wmma_f16(aq[c], bk, s);
      }
      sacc[nt] = s;
    }

    // ---- online softmax (row stats replicated across each 16-lane half)
    const float scale = 0.08838834764831845f;  // 1/sqrt(128)
    const int qrow_base = qt * 128 + w * 16 + 8 * (lane >> 4);
    const int key_base  = jt * 128 + (lane & 15);
#pragma unroll
    for (int r = 0; r < 8; ++r) {
      const int qrow = qrow_base + r;
      float vmax = -1e30f;
#pragma unroll
      for (int nt = 0; nt < 8; ++nt) {
        const int key = key_base + nt * 16;
        float s = sacc[nt][r] * scale;
        s = (key > qrow) ? -1e30f : s;   // causal mask
        sacc[nt][r] = s;
        vmax = fmaxf(vmax, s);
      }
#pragma unroll
      for (int msk = 1; msk < 16; msk <<= 1)
        vmax = fmaxf(vmax, __shfl_xor(vmax, msk, 32));
      const float mnew = fmaxf(mrow[r], vmax);
      const float corr = __expf(mrow[r] - mnew);
      float rsum = 0.0f;
#pragma unroll
      for (int nt = 0; nt < 8; ++nt) {
        const float p = __expf(sacc[nt][r] - mnew);
        sacc[nt][r] = p;
        rsum += p;
      }
#pragma unroll
      for (int msk = 1; msk < 16; msk <<= 1)
        rsum += __shfl_xor(rsum, msk, 32);
      lrow[r] = lrow[r] * corr + rsum;
      mrow[r] = mnew;
#pragma unroll
      for (int dt = 0; dt < 8; ++dt) o[dt][r] *= corr;
    }

    __syncthreads();  // all waves finished reading kls (B-frags)
    // ---- C-layout -> A-layout via per-wave LDS region (reuses kls)
    half_t* pA = kls + w * (16 * 128);
    {
      const int rowoff = 8 * (lane >> 4);
#pragma unroll
      for (int r = 0; r < 8; ++r)
#pragma unroll
        for (int nt = 0; nt < 8; ++nt)
          pA[(rowoff + r) * 128 + nt * 16 + (lane & 15)] = (half_t)sacc[nt][r];
    }
    // wave-local LDS RAW: drain our DS stores before DS fragment reads
    asm volatile("s_wait_dscnt 0" ::: "memory");

    // ---- O += P @ V
#pragma unroll
    for (int c = 0; c < 4; ++c) {
      v16h ap = load_afrag(pA + c * 32, 128, lane);
#pragma unroll
      for (int dt = 0; dt < 8; ++dt) {
        v16h bv = load_bfrag(vls + (dt * 16) * 128 + c * 32, 128, lane);
        o[dt] = wmma_f16(ap, bv, o[dt]);
      }
    }
  }

  // ---- normalize and store context f16 as [b][t][h*dk + d]
  const size_t ctx_base = ((size_t)b * T_) * D_ + (size_t)h * DK_;
#pragma unroll
  for (int r = 0; r < 8; ++r) {
    const float inv = 1.0f / lrow[r];
    const int trow = qt * 128 + w * 16 + 8 * (lane >> 4) + r;
#pragma unroll
    for (int dt = 0; dt < 8; ++dt) {
      const float val = o[dt][r] * inv;
      ctx[ctx_base + (size_t)trow * D_ + dt * 16 + (lane & 15)] = (half_t)val;
    }
  }
}

// ---------------------------------------------------------------------------
// Kernel 3: out = ctx @ Wo + bo (fp32 out). grid (D/128, B*T/128), block 256.
// A tile is pure f16 copy -> async DMA to LDS when available.
// ---------------------------------------------------------------------------
__global__ __launch_bounds__(256) void out_proj_kernel(
    const half_t* __restrict__ ctx, const float* __restrict__ Wo,
    const float* __restrict__ bo, float* __restrict__ out) {
  __shared__ half_t As[128 * 32];
  __shared__ half_t Bs[128 * 32];   // transposed [n][k]

  const int n0 = blockIdx.x * 128;
  const int m0 = blockIdx.y * 128;
  const int tid = threadIdx.x, lane = tid & 31, w = tid >> 5;
  const int wm = (w & 3) * 32;
  const int wn = (w >> 2) * 64;

  v8f acc[2][4];
#pragma unroll
  for (int i = 0; i < 2; ++i)
#pragma unroll
    for (int j = 0; j < 4; ++j) acc[i][j] = zero8();

  const int arow = tid >> 1, ac0 = (tid & 1) * 16;
  const int brow = tid >> 3, bc0 = (tid & 7) * 16;

  for (int kk = 0; kk < D_; kk += 32) {
    {  // A: already f16 -> async copy to LDS
      const half_t* src = ctx + (size_t)(m0 + arow) * D_ + kk + ac0;
      half_t* dst = As + arow * 32 + ac0;
#ifdef HAVE_ASYNC_LDS
      async_copy16B(src, dst);
      async_copy16B(src + 8, dst + 8);
#else
      ((v8h*)dst)[0] = ((const v8h*)src)[0];
      ((v8h*)dst)[1] = ((const v8h*)src)[1];
#endif
    }
    {  // B: fp32 Wo -> f16, transposed
      const float4* src = (const float4*)(Wo + (size_t)(kk + brow) * D_ + n0 + bc0);
#pragma unroll
      for (int i = 0; i < 4; ++i) {
        float4 v = src[i];
        Bs[(bc0 + i * 4 + 0) * 32 + brow] = (half_t)v.x;
        Bs[(bc0 + i * 4 + 1) * 32 + brow] = (half_t)v.y;
        Bs[(bc0 + i * 4 + 2) * 32 + brow] = (half_t)v.z;
        Bs[(bc0 + i * 4 + 3) * 32 + brow] = (half_t)v.w;
      }
    }
    if (kk + 32 < D_) {
      __builtin_prefetch(ctx + (size_t)(m0 + arow) * D_ + kk + 32 + ac0, 0, 1);
      __builtin_prefetch(Wo + (size_t)(kk + 32 + brow) * D_ + n0 + bc0, 0, 1);
    }
#ifdef HAVE_ASYNC_LDS
    async_wait0();
#endif
    __syncthreads();
    v16h a0 = load_afrag(As + (wm +  0) * 32, 32, lane);
    v16h a1 = load_afrag(As + (wm + 16) * 32, 32, lane);
#pragma unroll
    for (int nt = 0; nt < 4; ++nt) {
      v16h b = load_bfrag(Bs + (wn + nt * 16) * 32, 32, lane);
      acc[0][nt] = wmma_f16(a0, b, acc[0][nt]);
      acc[1][nt] = wmma_f16(a1, b, acc[1][nt]);
    }
    __syncthreads();
  }

#pragma unroll
  for (int mt = 0; mt < 2; ++mt) {
#pragma unroll
    for (int nt = 0; nt < 4; ++nt) {
      const int n_g = n0 + wn + nt * 16 + (lane & 15);
      const float bias = bo[n_g];
      const int m_base = m0 + wm + mt * 16 + 8 * (lane >> 4);
#pragma unroll
      for (int r = 0; r < 8; ++r)
        out[(size_t)(m_base + r) * D_ + n_g] = acc[mt][nt][r] + bias;
    }
  }
}

// ---------------------------------------------------------------------------
extern "C" void kernel_launch(void* const* d_in, const int* in_sizes, int n_in,
                              void* d_out, int out_size, void* d_ws,
                              size_t ws_size, hipStream_t stream) {
  const float* x    = (const float*)d_in[0];
  const float* Wkqv = (const float*)d_in[1];
  const float* bkqv = (const float*)d_in[2];
  const float* Wo   = (const float*)d_in[3];
  const float* bo   = (const float*)d_in[4];
  float* out = (float*)d_out;

  half_t* qkv = (half_t*)d_ws;          // 3*SEC_ halves (48 MB)
  half_t* ctx = qkv + 3 * SEC_;         // B*T*D halves  (16 MB)

  qkv_rope_kernel<<<dim3(ND_ / 128, (B_ * T_) / 128), 256, 0, stream>>>(
      x, Wkqv, bkqv, qkv);
  attn_kernel<<<dim3(T_ / 128, B_ * H_), 256, 0, stream>>>(qkv, ctx);
  out_proj_kernel<<<dim3(D_ / 128, (B_ * T_) / 128), 256, 0, stream>>>(
      ctx, Wo, bo, out);
}
